// Attention_7602092114471
// MI455X (gfx1250) — compile-verified, matrix operand order checked by reading, not run on hardware
//
#include <hip/hip_runtime.h>

// ---------- types ----------
typedef __bf16 bf16;
typedef __attribute__((ext_vector_type(16))) __bf16        v16bf;
typedef __attribute__((ext_vector_type(8)))  float         v8f;
typedef __attribute__((ext_vector_type(4)))  unsigned int  u32x4;

union Frag { v16bf v; u32x4 q[2]; };   // 16 bf16 = 8 VGPRs, loaded as 2x16B

// ---------- problem constants ----------
#define BB   2
#define NN   2048
#define CC   768
#define HH   12
#define DD   64
#define MM   (BB*NN)        // 4096 token rows
#define QKVC (3*CC)         // 2304

#define WMMA_BF16(A, B, C) \
    __builtin_amdgcn_wmma_f32_16x16x32_bf16(false, (A), false, (B), (short)0, (C), false, false)

// A/B fragment: lane (lrow = lane&15) owns row/col, khalf = lane>>4 selects the
// K-halves; 16 bf16 elements = two contiguous 8-elem runs at +khalf*8 and
// +16+khalf*8 (CDNA5 ISA 16-bit A/B VGPR layout).
static __device__ __forceinline__ void load_frag(Frag& f, const bf16* base_plus_khalf8) {
    f.q[0] = *(const u32x4*)(base_plus_khalf8);
    f.q[1] = *(const u32x4*)(base_plus_khalf8 + 16);
}

// ---------- kernel 0: fp32 -> bf16 convert ----------
__global__ __launch_bounds__(256) void cvt_bf16(const float* __restrict__ s,
                                                bf16* __restrict__ d, int n) {
    int i = blockIdx.x * 256 + threadIdx.x;
    if (i < n) d[i] = (bf16)s[i];
}

// ---------- kernel 1: QKV GEMM (x @ Wqkv^T) fused with scale + RoPE ----------
// One wave computes a 32x32 block (2x2 WMMA tiles) of the [4096 x 2304] qkv
// matrix: 2 A-frags + 2 B-frags -> 4 WMMAs per k-step (2x data reuse).
// Writes: qb,kb as [B,H,N,64] bf16 (RoPE'd, q scaled), vt as [B,H,64,N] bf16.
__global__ __launch_bounds__(256) void qkv_rope(const bf16* __restrict__ xb,
                                                const bf16* __restrict__ wb,
                                                bf16* __restrict__ qb,
                                                bf16* __restrict__ kb,
                                                bf16* __restrict__ vt) {
    const int lane  = threadIdx.x & 31;
    const int wid   = threadIdx.x >> 5;
    const int w     = blockIdx.x * 8 + wid;
    const int tn    = w % (QKVC / 32);     // 72 col-blocks
    const int tm    = w / (QKVC / 32);     // 128 row-blocks
    const int lrow  = lane & 15;
    const int khalf = lane >> 4;

    const bf16* arow0 = xb + (size_t)(tm * 32 + lrow) * CC + khalf * 8;
    const bf16* arow1 = arow0 + (size_t)16 * CC;
    const bf16* brow0 = wb + (size_t)(tn * 32 + lrow) * CC + khalf * 8;
    const bf16* brow1 = brow0 + (size_t)16 * CC;

    v8f acc[2][2] = {};
    #pragma unroll 2
    for (int kk = 0; kk < CC; kk += 32) {
        Frag a0, a1, b0, b1;
        load_frag(a0, arow0 + kk);
        load_frag(a1, arow1 + kk);
        load_frag(b0, brow0 + kk);
        load_frag(b1, brow1 + kk);
        acc[0][0] = WMMA_BF16(a0.v, b0.v, acc[0][0]);
        acc[0][1] = WMMA_BF16(a0.v, b1.v, acc[0][1]);
        acc[1][0] = WMMA_BF16(a1.v, b0.v, acc[1][0]);
        acc[1][1] = WMMA_BF16(a1.v, b1.v, acc[1][1]);
    }

    // C/D layout: lane owns col n = lane&15; VGPR r holds row r + khalf*8.
    const int seg = (tn * 32) / CC;        // 0=q,1=k,2=v (32-block never straddles)
    #pragma unroll
    for (int nj = 0; nj < 2; ++nj) {
        const int ncol = tn * 32 + nj * 16 + lrow;
        const int dcol = ncol & 63;
        if (seg == 2) {
            const int h = (ncol - 2 * CC) >> 6;
            #pragma unroll
            for (int mi = 0; mi < 2; ++mi) {
                #pragma unroll
                for (int r = 0; r < 8; ++r) {
                    int m = tm * 32 + mi * 16 + r + khalf * 8;
                    int b = m >> 11, pos = m & (NN - 1);
                    vt[((size_t)(b * HH + h) * DD + dcol) * NN + pos] = (bf16)acc[mi][nj][r];
                }
            }
        } else {
            const float scale = (seg == 0) ? 0.125f : 1.0f;   // d^-0.5 on q only
            const int   h     = (ncol - seg * CC) >> 6;
            // interleaved RoPE: inv_freq = theta^(-(dcol&~1)/64); pair = adjacent col
            const float invf  = __powf(10000.0f, -(float)(dcol & ~1) * (1.0f / 64.0f));
            bf16* dst = (seg == 0) ? qb : kb;
            #pragma unroll
            for (int mi = 0; mi < 2; ++mi) {
                #pragma unroll
                for (int r = 0; r < 8; ++r) {
                    int   m   = tm * 32 + mi * 16 + r + khalf * 8;
                    int   b   = m >> 11, pos = m & (NN - 1);
                    float val = acc[mi][nj][r] * scale;
                    float pv  = __shfl_xor(val, 1, 32);       // partner column n^1
                    float rot = (dcol & 1) ? pv : -pv;        // (-t_odd, t_even)
                    float sa, ca;
                    __sincosf((float)pos * invf, &sa, &ca);
                    dst[((size_t)(b * HH + h) * NN + pos) * DD + dcol] =
                        (bf16)(val * ca + rot * sa);
                }
            }
        }
    }
}

// ---------- kernel 2: flash attention ------------------------------------
// Block = 8 waves = 128 query rows of one (b,h). Each 32-key K/V tile is
// staged ONCE per block into LDS (8x traffic cut vs per-wave streaming);
// waves build WMMA fragments with ds_load_b128. Next tile is prefetched.
__global__ __launch_bounds__(256) void attn_flash(const bf16* __restrict__ qb,
                                                  const bf16* __restrict__ kb,
                                                  const bf16* __restrict__ vt,
                                                  bf16* __restrict__ ob) {
    __shared__ __align__(16) bf16 kt [32 * 64];      // 32 keys x 64 d
    __shared__ __align__(16) bf16 vts[64 * 32];      // 64 d x 32 j (V^T layout)
    __shared__ __align__(16) bf16 psh[8][16 * 32];   // per-wave P staging
    const int tid   = threadIdx.x;
    const int lane  = tid & 31;
    const int wid   = tid >> 5;
    const int bh    = blockIdx.x >> 4;               // 24 (b,h), 16 blocks each
    const int iblk  = blockIdx.x & 15;
    const int lrow  = lane & 15;
    const int khalf = lane >> 4;

    const bf16* qh = qb + (size_t)bh * NN * DD;
    const bf16* kh = kb + (size_t)bh * NN * DD;
    const bf16* vh = vt + (size_t)bh * DD * NN;
    bf16* pb = &psh[wid][0];

    // Q fragments for the two d-halves (d=64 -> K-chunks 0..31 and 32..63)
    const int irow = iblk * 128 + wid * 16;
    Frag qf0, qf1;
    const bf16* qrow = qh + (size_t)(irow + lrow) * DD + khalf * 8;
    load_frag(qf0, qrow);
    load_frag(qf1, qrow + 32);

    // staging assignments: 256 threads x one 16B chunk each
    const int krow_s = tid >> 3, kchk = (tid & 7) * 8;   // 32 rows x 8 chunks
    const int vrow_s = tid >> 2, vchk = (tid & 3) * 8;   // 64 rows x 4 chunks

    v8f   o[4] = {};
    float mrow[8], lsum[8];
    #pragma unroll
    for (int r = 0; r < 8; ++r) { mrow[r] = -1e30f; lsum[r] = 0.0f; }

    for (int j0 = 0; j0 < NN; j0 += 32) {
        // ---- stage K/V tile into LDS (one 16B chunk per thread) ----
        *(u32x4*)&kt [krow_s * 64 + kchk] =
            *(const u32x4*)(kh + (size_t)(j0 + krow_s) * DD + kchk);
        *(u32x4*)&vts[vrow_s * 32 + vchk] =
            *(const u32x4*)(vh + (size_t)vrow_s * NN + j0 + vchk);
        if (j0 + 32 < NN) {   // prefetch next tile into cache
            __builtin_prefetch((const void*)(kh + (size_t)(j0 + 32 + krow_s) * DD + kchk), 0, 0);
            __builtin_prefetch((const void*)(vh + (size_t)vrow_s * NN + j0 + 32 + vchk), 0, 0);
        }
        __syncthreads();

        // ---- S = Q K^T for 32 keys (two 16-col subtiles) from LDS ----
        v8f s0 = {}, s1 = {};
        {
            Frag kf;
            load_frag(kf, &kt[(lrow) * 64 + khalf * 8]);
            s0 = WMMA_BF16(qf0.v, kf.v, s0);
            load_frag(kf, &kt[(lrow) * 64 + 32 + khalf * 8]);
            s0 = WMMA_BF16(qf1.v, kf.v, s0);
            load_frag(kf, &kt[(16 + lrow) * 64 + khalf * 8]);
            s1 = WMMA_BF16(qf0.v, kf.v, s1);
            load_frag(kf, &kt[(16 + lrow) * 64 + 32 + khalf * 8]);
            s1 = WMMA_BF16(qf1.v, kf.v, s1);
        }

        // ---- online softmax; row r+khalf*8 lives in this lane-half ----
        #pragma unroll
        for (int r = 0; r < 8; ++r) {
            float mx = fmaxf(s0[r], s1[r]);
            mx = fmaxf(mx, __shfl_xor(mx, 1, 32));
            mx = fmaxf(mx, __shfl_xor(mx, 2, 32));
            mx = fmaxf(mx, __shfl_xor(mx, 4, 32));
            mx = fmaxf(mx, __shfl_xor(mx, 8, 32));
            float nm = fmaxf(mrow[r], mx);
            float a  = __expf(mrow[r] - nm);
            float p0 = __expf(s0[r] - nm);
            float p1 = __expf(s1[r] - nm);
            float rs = p0 + p1;
            rs += __shfl_xor(rs, 1, 32);
            rs += __shfl_xor(rs, 2, 32);
            rs += __shfl_xor(rs, 4, 32);
            rs += __shfl_xor(rs, 8, 32);
            lsum[r] = lsum[r] * a + rs;
            mrow[r] = nm;
            o[0][r] *= a; o[1][r] *= a; o[2][r] *= a; o[3][r] *= a;
            // stage P (C-layout -> row-major LDS)
            pb[(r + khalf * 8) * 32 + lrow]      = (bf16)p0;
            pb[(r + khalf * 8) * 32 + 16 + lrow] = (bf16)p1;
        }
        asm volatile("s_wait_dscnt 0" ::: "memory");   // same-wave LDS RAW

        // ---- P as A-fragment, accumulate O += P @ V (V from LDS) ----
        Frag pf;
        pf.q[0] = *(const u32x4*)(pb + lrow * 32 + khalf * 8);
        pf.q[1] = *(const u32x4*)(pb + lrow * 32 + 16 + khalf * 8);
        #pragma unroll
        for (int c = 0; c < 4; ++c) {
            Frag vf;
            load_frag(vf, &vts[(c * 16 + lrow) * 32 + khalf * 8]);
            o[c] = WMMA_BF16(pf.v, vf.v, o[c]);
        }
        __syncthreads();   // protect kt/vts before next stage
    }

    // ---- normalize and store as bf16 [B,N,C] for the proj GEMM ----
    const int b = bh / HH, h = bh % HH;
    #pragma unroll
    for (int r = 0; r < 8; ++r) {
        float inv = 1.0f / lsum[r];
        int pos = irow + r + khalf * 8;
        size_t rowoff = ((size_t)(b * NN + pos)) * CC + h * DD + lrow;
        ob[rowoff +  0] = (bf16)(o[0][r] * inv);
        ob[rowoff + 16] = (bf16)(o[1][r] * inv);
        ob[rowoff + 32] = (bf16)(o[2][r] * inv);
        ob[rowoff + 48] = (bf16)(o[3][r] * inv);
    }
}

// ---------- kernel 3: proj GEMM (attn_out @ Wproj^T + b) -> f32 out ----------
// 32x32 block per wave (2x2 WMMA tiles), same reuse scheme as kernel 1.
__global__ __launch_bounds__(256) void proj_gemm(const bf16* __restrict__ ab,
                                                 const bf16* __restrict__ wpb,
                                                 const float* __restrict__ bias,
                                                 float* __restrict__ out) {
    const int lane  = threadIdx.x & 31;
    const int wid   = threadIdx.x >> 5;
    const int w     = blockIdx.x * 8 + wid;
    const int tn    = w % (CC / 32);       // 24 col-blocks
    const int tm    = w / (CC / 32);       // 128 row-blocks
    const int lrow  = lane & 15;
    const int khalf = lane >> 4;

    const bf16* arow0 = ab  + (size_t)(tm * 32 + lrow) * CC + khalf * 8;
    const bf16* arow1 = arow0 + (size_t)16 * CC;
    const bf16* brow0 = wpb + (size_t)(tn * 32 + lrow) * CC + khalf * 8;
    const bf16* brow1 = brow0 + (size_t)16 * CC;

    v8f acc[2][2] = {};
    #pragma unroll 2
    for (int kk = 0; kk < CC; kk += 32) {
        Frag a0, a1, b0, b1;
        load_frag(a0, arow0 + kk);
        load_frag(a1, arow1 + kk);
        load_frag(b0, brow0 + kk);
        load_frag(b1, brow1 + kk);
        acc[0][0] = WMMA_BF16(a0.v, b0.v, acc[0][0]);
        acc[0][1] = WMMA_BF16(a0.v, b1.v, acc[0][1]);
        acc[1][0] = WMMA_BF16(a1.v, b0.v, acc[1][0]);
        acc[1][1] = WMMA_BF16(a1.v, b1.v, acc[1][1]);
    }

    #pragma unroll
    for (int nj = 0; nj < 2; ++nj) {
        const int   n  = tn * 32 + nj * 16 + lrow;
        const float bn = bias[n];
        #pragma unroll
        for (int mi = 0; mi < 2; ++mi) {
            #pragma unroll
            for (int r = 0; r < 8; ++r) {
                int m = tm * 32 + mi * 16 + r + khalf * 8;
                out[(size_t)m * CC + n] = acc[mi][nj][r] + bn;
            }
        }
    }
}

// ---------- launcher ----------
extern "C" void kernel_launch(void* const* d_in, const int* in_sizes, int n_in,
                              void* d_out, int out_size, void* d_ws, size_t ws_size,
                              hipStream_t stream) {
    (void)in_sizes; (void)n_in; (void)out_size; (void)ws_size;
    const float* x     = (const float*)d_in[0];   // [2,2048,768]
    const float* wqkv  = (const float*)d_in[1];   // [2304,768]
    const float* wproj = (const float*)d_in[2];   // [768,768]
    const float* bproj = (const float*)d_in[3];   // [768]

    char*  ws  = (char*)d_ws;
    size_t off = 0;
    auto carve = [&](size_t elems) -> bf16* {
        bf16* p = (bf16*)(ws + off);
        off += ((elems * sizeof(bf16) + 255) / 256) * 256;
        return p;
    };
    bf16* xb    = carve((size_t)MM * CC);
    bf16* wqkvb = carve((size_t)QKVC * CC);
    bf16* wprojb= carve((size_t)CC * CC);
    bf16* qb    = carve((size_t)BB * HH * NN * DD);
    bf16* kbuf  = carve((size_t)BB * HH * NN * DD);
    bf16* vtb   = carve((size_t)BB * HH * DD * NN);
    bf16* attnb = carve((size_t)MM * CC);

    cvt_bf16<<<(MM * CC + 255) / 256,   256, 0, stream>>>(x,     xb,     MM * CC);
    cvt_bf16<<<(QKVC * CC + 255) / 256, 256, 0, stream>>>(wqkv,  wqkvb,  QKVC * CC);
    cvt_bf16<<<(CC * CC + 255) / 256,   256, 0, stream>>>(wproj, wprojb, CC * CC);

    // (4096/32)*(2304/32) = 9216 waves / 8 waves-per-block
    qkv_rope<<<1152, 256, 0, stream>>>(xb, wqkvb, qb, kbuf, vtb);
    // 24 (b,h) * 16 row-blocks
    attn_flash<<<384, 256, 0, stream>>>(qb, kbuf, vtb, attnb);
    // (4096/32)*(768/32) = 3072 waves / 8
    proj_gemm<<<384, 256, 0, stream>>>(attnb, wprojb, bproj, (float*)d_out);
}
